// Decoder_10230612099842
// MI455X (gfx1250) — matrix operand order repaired and checked by reading, not compiled, not run on hardware
//
#include <hip/hip_runtime.h>
#include <hip/hip_bf16.h>
#include <math.h>

// ---------------- problem constants ----------------
constexpr int Bc = 2, Kc = 512, Tc = 2048, Dc = 1024;
constexpr int NHc = 16, HDc = 64, DFFc = 4096, Lc = 2;
constexpr int MT = Bc * Tc;              // 4096 token rows
constexpr float EPSc = 1e-5f;

// ---------------- WMMA types / helpers ----------------
typedef __attribute__((ext_vector_type(16))) __bf16 v16bf;
typedef __attribute__((ext_vector_type(8)))  float  v8f;
typedef __attribute__((ext_vector_type(4)))  unsigned u32x4;
typedef __attribute__((ext_vector_type(8)))  int      i32x8;
typedef __attribute__((ext_vector_type(4)))  int      i32x4;

static __device__ __forceinline__ unsigned short f2bf(float f) {
  union { __bf16 h; unsigned short u; } cv;
  cv.h = (__bf16)f;                      // hw cvt, round-to-nearest-even
  return cv.u;
}
static __device__ __forceinline__ float bf2f(unsigned short h) {
  return __uint_as_float(((unsigned)h) << 16);
}
static __device__ __forceinline__ v16bf mkfrag(uint4 lo, uint4 hi) {
  union { uint4 u[2]; v16bf v; } x; x.u[0] = lo; x.u[1] = hi; return x.v;
}

// ---------------- gather (searchsorted) + residual ----------------
__global__ __launch_bounds__(256)
void k_gather_add(const float* __restrict__ xp, const int* __restrict__ bounds,
                  const float* __restrict__ xres, float* __restrict__ x) {
  int bt = blockIdx.x;
  int b = bt / Tc, t = bt % Tc;
  const int* row = bounds + (size_t)b * Kc;
  int lo = 0, hi = Kc;                    // searchsorted(row, t, side='right')
  while (lo < hi) { int mid = (lo + hi) >> 1; if (row[mid] <= t) lo = mid + 1; else hi = mid; }
  const float* src = xp + ((size_t)b * Kc + lo) * Dc;
  const float* res = xres + ((size_t)b * Tc + t) * Dc;
  float* dst = x + ((size_t)b * Tc + t) * Dc;
  for (int i = threadIdx.x; i < Dc; i += 256) dst[i] = src[i] + res[i];
}

// ---------------- RMSNorm (one block per row, 256 thr x 4 elems) ----------------
template <bool BF16OUT>
__global__ __launch_bounds__(256)
void k_rmsnorm(const float* __restrict__ X, const float* __restrict__ Wt,
               void* __restrict__ OutP) {
  int row = blockIdx.x;
  const float4 xv = ((const float4*)(X + (size_t)row * Dc))[threadIdx.x];
  float ss = xv.x * xv.x + xv.y * xv.y + xv.z * xv.z + xv.w * xv.w;
#pragma unroll
  for (int m = 1; m < 32; m <<= 1) ss += __shfl_xor(ss, m, 32);
  __shared__ float red[8];
  if ((threadIdx.x & 31) == 0) red[threadIdx.x >> 5] = ss;
  __syncthreads();
  float tot = 0.f;
#pragma unroll
  for (int i = 0; i < 8; i++) tot += red[i];
  float rinv = rsqrtf(tot * (1.0f / Dc) + EPSc);
  float4 w4 = ((const float4*)Wt)[threadIdx.x];
  float o0 = xv.x * rinv * w4.x, o1 = xv.y * rinv * w4.y;
  float o2 = xv.z * rinv * w4.z, o3 = xv.w * rinv * w4.w;
  if (BF16OUT) {
    uint2 pk;
    pk.x = (unsigned)f2bf(o0) | ((unsigned)f2bf(o1) << 16);
    pk.y = (unsigned)f2bf(o2) | ((unsigned)f2bf(o3) << 16);
    ((uint2*)OutP)[(size_t)row * (Dc / 4) + threadIdx.x] = pk;
  } else {
    float4 r; r.x = o0; r.y = o1; r.z = o2; r.w = o3;
    ((float4*)OutP)[(size_t)row * (Dc / 4) + threadIdx.x] = r;
  }
}

// ---------------- f32 -> bf16 bulk convert ----------------
__global__ __launch_bounds__(256)
void k_cvt_bf16(const float* __restrict__ in, unsigned short* __restrict__ out, int n4) {
  int i = blockIdx.x * 256 + threadIdx.x;
  if (i < n4) {
    float4 v = ((const float4*)in)[i];
    uint2 pk;
    pk.x = (unsigned)f2bf(v.x) | ((unsigned)f2bf(v.y) << 16);
    pk.y = (unsigned)f2bf(v.z) | ((unsigned)f2bf(v.w) << 16);
    ((uint2*)out)[i] = pk;
  }
}

// ---------------- RoPE + repack to [B,NH,T,HD] bf16 ----------------
__global__ __launch_bounds__(256)
void k_rope_pack(const float* __restrict__ q, const float* __restrict__ k,
                 const float* __restrict__ v, const float* __restrict__ cosb,
                 const float* __restrict__ sinb, unsigned short* __restrict__ qb,
                 unsigned short* __restrict__ kb, unsigned short* __restrict__ vb) {
  int gid = blockIdx.x * 256 + threadIdx.x;   // over B*T*NH*HD
  int hd = gid % HDc;
  int nh = (gid / HDc) % NHc;
  int t  = (gid / (HDc * NHc)) % Tc;
  int b  = gid / (HDc * NHc * Tc);
  size_t in_idx = ((size_t)(b * Tc + t) * NHc + nh) * HDc + hd;
  float c = cosb[t * HDc + hd], s = sinb[t * HDc + hd];
  int hd2 = (hd < HDc / 2) ? hd + HDc / 2 : hd - HDc / 2;
  float sign = (hd < HDc / 2) ? -1.f : 1.f;
  size_t rot_idx = in_idx + (hd2 - hd);
  float qo = q[in_idx] * c + sign * q[rot_idx] * s;
  float ko = k[in_idx] * c + sign * k[rot_idx] * s;
  size_t out_idx = ((size_t)(b * NHc + nh) * Tc + t) * HDc + hd;
  qb[out_idx] = f2bf(qo);
  kb[out_idx] = f2bf(ko);
  vb[out_idx] = f2bf(v[in_idx]);
}

// ---------------- tiled BF16 WMMA GEMM ----------------
// C[M,N] (op)= A[M,K](bf16, row-major) x W[K,N](f32, converted to bf16 in LDS)
// Block tile 128x128xBK32, 256 threads = 8 waves, each wave 32x64 (2x4 WMMA tiles)
// A-tile staged via Tensor Data Mover (tensor_load_to_lds), W-tile staged per-thread
// (fuses fp32->bf16 conversion + transpose which TDM cannot do).
// MODE: 0 = C=val ; 1 = C+=val ; 2 = Ob=bf16(silu(val)) ; 3 = Ob=bf16(val*bf2f(Aux))
template <int MODE>
__global__ __launch_bounds__(256)
void k_gemm(const unsigned short* __restrict__ A, const float* __restrict__ W,
            float* __restrict__ C, unsigned short* __restrict__ Ob,
            const unsigned short* __restrict__ Aux, int Msz, int Ksz, int Nsz) {
  __shared__ unsigned short As[128 * 32];          // [row][k] bf16
  __shared__ unsigned short Bs[128 * 32];          // [n][k]  bf16 (transposed)
  const int t = threadIdx.x;
  const int m0 = blockIdx.y * 128, n0 = blockIdx.x * 128;
  const int lane = t & 31, wv = t >> 5;
  const int wr = wv >> 1, wc = wv & 1;             // 4x2 wave grid
  const int lrow = lane & 15, lhalf = lane >> 4;

  v8f acc[2][4];
#pragma unroll
  for (int i = 0; i < 2; i++)
#pragma unroll
    for (int j = 0; j < 4; j++) acc[i][j] = v8f{};

  for (int k0 = 0; k0 < Ksz; k0 += 32) {
    // --- stage A tile (bf16, 128 rows x 32 cols) via TDM ---
#if __has_builtin(__builtin_amdgcn_tensor_load_to_lds)
    if (t < 32) {                                  // wave 0 issues the DMA
      unsigned lds_a = (unsigned)(size_t)(&As[0]);
      unsigned long long ga =
          (unsigned long long)(size_t)A + ((size_t)m0 * Ksz + k0) * 2ull;
      u32x4 g0;
      g0.x = 1u;                                   // count=1, no gather/iterate
      g0.y = lds_a;                                // lds_addr (bytes)
      g0.z = (unsigned)(ga & 0xffffffffu);         // global_addr[31:0]
      g0.w = (unsigned)((ga >> 32) & 0x01ffffffu) | (2u << 30);  // addr[56:32]|type=2
      i32x8 g1;
      g1[0] = (int)(1u << 16);                     // data_size=1 (2 bytes/elem)
      g1[1] = (int)(((unsigned)Ksz & 0xffffu) << 16);            // tensor_dim0 lo16
      g1[2] = (int)(((unsigned)Ksz >> 16) | (((unsigned)Msz & 0xffffu) << 16));
      g1[3] = (int)(((unsigned)Msz >> 16) | (32u << 16));        // tile_dim0=32
      g1[4] = 128;                                 // tile_dim1=128, tile_dim2=0
      g1[5] = (int)(unsigned)Ksz;                  // tensor_dim0_stride lo32
      g1[6] = 0;                                   // stride hi, dim1_stride lo
      g1[7] = 0;
      i32x4 g2 = (i32x4){0, 0, 0, 0};
      i32x4 g3 = (i32x4){0, 0, 0, 0};
      i32x8 g4 = (i32x8){0, 0, 0, 0, 0, 0, 0, 0};
      __builtin_amdgcn_tensor_load_to_lds(g0, g1, g2, g3, g4, 0);
      __builtin_amdgcn_s_wait_tensorcnt(0);
    }
#else
    {
      int colb = (t & 3) * 8;
#pragma unroll
      for (int p = 0; p < 2; p++) {
        int row = (t >> 2) + p * 64;
        uint4 v = *(const uint4*)(A + (size_t)(m0 + row) * Ksz + k0 + colb);
        *(uint4*)&As[row * 32 + colb] = v;
      }
    }
#endif
    // --- stage W tile (f32 32x128 -> bf16 Bs[n][k]) ---
    {
      int nIdx = lane * 4;
#pragma unroll
      for (int p = 0; p < 4; p++) {
        int kr = (t >> 5) + p * 8;
        float4 w4 = *(const float4*)(W + (size_t)(k0 + kr) * Nsz + n0 + nIdx);
        Bs[(nIdx + 0) * 32 + kr] = f2bf(w4.x);
        Bs[(nIdx + 1) * 32 + kr] = f2bf(w4.y);
        Bs[(nIdx + 2) * 32 + kr] = f2bf(w4.z);
        Bs[(nIdx + 3) * 32 + kr] = f2bf(w4.w);
      }
    }
    if (k0 + 32 < Ksz) {                 // hint next W tile into cache
      __builtin_prefetch(W + (size_t)(k0 + 32 + (t >> 5)) * Nsz + n0 + lane * 4, 0, 1);
    }
    __syncthreads();

    // --- fragments + 8 WMMA per wave ---
    v16bf af[2];
#pragma unroll
    for (int mt = 0; mt < 2; mt++) {
      int row = wr * 32 + mt * 16 + lrow;
      uint4 lo = *(const uint4*)&As[row * 32 + lhalf * 8];
      uint4 hi = *(const uint4*)&As[row * 32 + 16 + lhalf * 8];
      af[mt] = mkfrag(lo, hi);
    }
#pragma unroll
    for (int nt = 0; nt < 4; nt++) {
      int col = wc * 64 + nt * 16 + lrow;
      uint4 lo = *(const uint4*)&Bs[col * 32 + lhalf * 8];
      uint4 hi = *(const uint4*)&Bs[col * 32 + 16 + lhalf * 8];
      v16bf bfr = mkfrag(lo, hi);
#pragma unroll
      for (int mt = 0; mt < 2; mt++)
        acc[mt][nt] = __builtin_amdgcn_wmma_f32_16x16x32_bf16(
            false, af[mt], false, bfr, (short)0, acc[mt][nt], false, false);
    }
    __syncthreads();
  }

  // --- epilogue ---
#pragma unroll
  for (int mt = 0; mt < 2; mt++)
#pragma unroll
    for (int nt = 0; nt < 4; nt++) {
      int colg = n0 + wc * 64 + nt * 16 + lrow;
#pragma unroll
      for (int v = 0; v < 8; v++) {
        int rowg = m0 + wr * 32 + mt * 16 + v + lhalf * 8;
        size_t idx = (size_t)rowg * Nsz + colg;
        float val = acc[mt][nt][v];
        if (MODE == 0) {
          C[idx] = val;
        } else if (MODE == 1) {
          C[idx] += val;
        } else if (MODE == 2) {
          Ob[idx] = f2bf(val / (1.f + __expf(-val)));      // silu
        } else {
          Ob[idx] = f2bf(val * bf2f(Aux[idx]));            // gate multiply
        }
      }
    }
}

// ---------------- flash attention: 1 wave per 16-query tile ----------------
// q,k,v: bf16 [B,NH,T,HD]; out: f32 [B,T,D] (heads merged)
__global__ __launch_bounds__(128)
void k_attn(const unsigned short* __restrict__ Q, const unsigned short* __restrict__ Kb,
            const unsigned short* __restrict__ V, float* __restrict__ Out) {
  __shared__ unsigned short Pl[4][16 * 32];     // probs tile per wave
  __shared__ unsigned short Vt[4][64 * 40];     // V chunk transposed [hd][key], pad 40
  const int lane = threadIdx.x & 31, wv = threadIdx.x >> 5;
  const int grpPerHead = Tc / 64;               // 4 q-tiles per block
  int bh  = blockIdx.x / grpPerHead;
  int grp = blockIdx.x % grpPerHead;
  int qbase = (grp * 4 + wv) * 16;
  int b = bh / NHc, h = bh % NHc;
  const int lrow = lane & 15, lhalf = lane >> 4;
  const float scale = 0.125f;                   // 1/sqrt(64)

  // Q fragments (A-layout straight from global, [T,HD] contiguous)
  const unsigned short* qp = Q + ((size_t)bh * Tc + qbase) * HDc;
  v16bf qf[2];
#pragma unroll
  for (int c = 0; c < 2; c++) {
    const unsigned short* r = qp + (size_t)lrow * HDc + c * 32 + lhalf * 8;
    qf[c] = mkfrag(*(const uint4*)r, *(const uint4*)(r + 16));
  }

  float mrun[8], lrun[8];
  v8f oa[4];
#pragma unroll
  for (int v = 0; v < 8; v++) { mrun[v] = -1e30f; lrun[v] = 0.f; }
#pragma unroll
  for (int ht = 0; ht < 4; ht++) oa[ht] = v8f{};

  int nch = (qbase + 16 + 31) >> 5;             // 32-key chunks (causal bound)
  for (int ch = 0; ch < nch; ch++) {
    int kb0 = ch * 32;
    // stage V chunk transposed: lane owns key kb0+lane
    {
      const uint4* vr = (const uint4*)(V + ((size_t)bh * Tc + kb0 + lane) * HDc);
#pragma unroll
      for (int i = 0; i < 8; i++) {
        uint4 r = vr[i];
        int hb = i * 8;
        Vt[wv][(hb + 0) * 40 + lane] = (unsigned short)(r.x & 0xffff);
        Vt[wv][(hb + 1) * 40 + lane] = (unsigned short)(r.x >> 16);
        Vt[wv][(hb + 2) * 40 + lane] = (unsigned short)(r.y & 0xffff);
        Vt[wv][(hb + 3) * 40 + lane] = (unsigned short)(r.y >> 16);
        Vt[wv][(hb + 4) * 40 + lane] = (unsigned short)(r.z & 0xffff);
        Vt[wv][(hb + 5) * 40 + lane] = (unsigned short)(r.z >> 16);
        Vt[wv][(hb + 6) * 40 + lane] = (unsigned short)(r.w & 0xffff);
        Vt[wv][(hb + 7) * 40 + lane] = (unsigned short)(r.w >> 16);
      }
    }
    // S = Q x K^T  (K B-fragments straight from global)
    v8f sc[2];
#pragma unroll
    for (int tI = 0; tI < 2; tI++) {
      const unsigned short* kp = Kb + ((size_t)bh * Tc + kb0 + tI * 16 + lrow) * HDc;
      v8f s = v8f{};
#pragma unroll
      for (int c = 0; c < 2; c++) {
        const unsigned short* r = kp + c * 32 + lhalf * 8;
        s = __builtin_amdgcn_wmma_f32_16x16x32_bf16(
            false, qf[c], false, mkfrag(*(const uint4*)r, *(const uint4*)(r + 16)),
            (short)0, s, false, false);
      }
      int key = kb0 + tI * 16 + lrow;
#pragma unroll
      for (int v = 0; v < 8; v++) {
        int qrow = qbase + v + lhalf * 8;
        float xv = s[v] * scale;
        s[v] = (key <= qrow) ? xv : -1e30f;     // causal mask
      }
      sc[tI] = s;
    }
    // online softmax (row reductions across 16-lane halves)
    float alpha[8];
#pragma unroll
    for (int v = 0; v < 8; v++) {
      float rm = fmaxf(sc[0][v], sc[1][v]);
#pragma unroll
      for (int m = 1; m < 16; m <<= 1) rm = fmaxf(rm, __shfl_xor(rm, m, 32));
      float mn = fmaxf(mrun[v], rm);
      alpha[v] = __expf(mrun[v] - mn);
      mrun[v] = mn;
    }
#pragma unroll
    for (int tI = 0; tI < 2; tI++)
#pragma unroll
      for (int v = 0; v < 8; v++) {
        float p = __expf(sc[tI][v] - mrun[v]);
        sc[tI][v] = p;
        Pl[wv][(v + lhalf * 8) * 32 + tI * 16 + lrow] = f2bf(p);
      }
#pragma unroll
    for (int v = 0; v < 8; v++) {
      float rs = sc[0][v] + sc[1][v];
#pragma unroll
      for (int m = 1; m < 16; m <<= 1) rs += __shfl_xor(rs, m, 32);
      lrun[v] = lrun[v] * alpha[v] + rs;
#pragma unroll
      for (int ht = 0; ht < 4; ht++) oa[ht][v] *= alpha[v];
    }
    // make P / Vt LDS writes visible within the wave (in-order DS pipe + wait)
    asm volatile("s_wait_dscnt 0" ::: "memory");
    __builtin_amdgcn_wave_barrier();
    // O += P x V
    uint4 plo = *(const uint4*)&Pl[wv][lrow * 32 + lhalf * 8];
    uint4 phi = *(const uint4*)&Pl[wv][lrow * 32 + 16 + lhalf * 8];
    v16bf pf = mkfrag(plo, phi);
#pragma unroll
    for (int ht = 0; ht < 4; ht++) {
      int hdc = ht * 16 + lrow;
      uint4 lo = *(const uint4*)&Vt[wv][hdc * 40 + lhalf * 8];
      uint4 hi = *(const uint4*)&Vt[wv][hdc * 40 + 16 + lhalf * 8];
      oa[ht] = __builtin_amdgcn_wmma_f32_16x16x32_bf16(
          false, pf, false, mkfrag(lo, hi), (short)0, oa[ht], false, false);
    }
  }
  // write O (merge heads into [B,T,D])
#pragma unroll
  for (int v = 0; v < 8; v++) {
    float inv = 1.f / lrun[v];
    int qrow = qbase + v + lhalf * 8;
    float* op = Out + ((size_t)b * Tc + qrow) * Dc + h * HDc;
#pragma unroll
    for (int ht = 0; ht < 4; ht++) op[ht * 16 + lrow] = oa[ht][v] * inv;
  }
}

// ---------------- host orchestration ----------------
extern "C" void kernel_launch(void* const* d_in, const int* in_sizes, int n_in,
                              void* d_out, int out_size, void* d_ws, size_t ws_size,
                              hipStream_t stream) {
  const float* xp      = (const float*)d_in[0];
  const int*   bounds  = (const int*)d_in[1];
  const float* xres    = (const float*)d_in[3];
  const float* cosb    = (const float*)d_in[4];
  const float* sinb    = (const float*)d_in[5];
  const float* wq      = (const float*)d_in[7];
  const float* wk      = (const float*)d_in[8];
  const float* wvw     = (const float*)d_in[9];
  const float* wo      = (const float*)d_in[10];
  const float* attn_nw = (const float*)d_in[11];
  const float* mlp_nw  = (const float*)d_in[12];
  const float* w1      = (const float*)d_in[13];
  const float* w3      = (const float*)d_in[14];
  const float* w2      = (const float*)d_in[15];
  const float* fin_nw  = (const float*)d_in[16];

  char* ws = (char*)d_ws;
  size_t off = 0;
  auto alloc = [&](size_t bytes) -> void* {
    void* p = ws + off; off += (bytes + 255) & ~(size_t)255; return p;
  };
  float* x            = (float*)alloc((size_t)MT * Dc * 4);
  unsigned short* hbf = (unsigned short*)alloc((size_t)MT * Dc * 2);
  float* qf           = (float*)alloc((size_t)MT * Dc * 4);
  float* kf           = (float*)alloc((size_t)MT * Dc * 4);
  float* vf           = (float*)alloc((size_t)MT * Dc * 4);
  unsigned short* qb  = (unsigned short*)alloc((size_t)MT * Dc * 2);
  unsigned short* kb2 = (unsigned short*)alloc((size_t)MT * Dc * 2);
  unsigned short* vb2 = (unsigned short*)alloc((size_t)MT * Dc * 2);
  float* ao           = (float*)alloc((size_t)MT * Dc * 4);
  unsigned short* aob = (unsigned short*)alloc((size_t)MT * Dc * 2);
  unsigned short* sbf = (unsigned short*)alloc((size_t)MT * DFFc * 2);
  unsigned short* gbf = (unsigned short*)alloc((size_t)MT * DFFc * 2);

  dim3 b256(256);
  k_gather_add<<<dim3(Bc * Tc), b256, 0, stream>>>(xp, bounds, xres, x);

  auto gemm = [&](int mode, const unsigned short* A, const float* W, float* C,
                  unsigned short* Ob, const unsigned short* Aux, int Ms, int Ks, int Ns) {
    dim3 g(Ns / 128, Ms / 128);
    switch (mode) {
      case 0: k_gemm<0><<<g, b256, 0, stream>>>(A, W, C, Ob, Aux, Ms, Ks, Ns); break;
      case 1: k_gemm<1><<<g, b256, 0, stream>>>(A, W, C, Ob, Aux, Ms, Ks, Ns); break;
      case 2: k_gemm<2><<<g, b256, 0, stream>>>(A, W, C, Ob, Aux, Ms, Ks, Ns); break;
      default: k_gemm<3><<<g, b256, 0, stream>>>(A, W, C, Ob, Aux, Ms, Ks, Ns); break;
    }
  };

  for (int l = 0; l < Lc; l++) {
    const float* wql = wq  + (size_t)l * Dc * Dc;
    const float* wkl = wk  + (size_t)l * Dc * Dc;
    const float* wvl = wvw + (size_t)l * Dc * Dc;
    const float* wol = wo  + (size_t)l * Dc * Dc;
    const float* w1l = w1  + (size_t)l * Dc * DFFc;
    const float* w3l = w3  + (size_t)l * Dc * DFFc;
    const float* w2l = w2  + (size_t)l * DFFc * Dc;

    k_rmsnorm<true><<<dim3(MT), b256, 0, stream>>>(x, attn_nw + (size_t)l * Dc, hbf);
    gemm(0, hbf, wql, qf, nullptr, nullptr, MT, Dc, Dc);
    gemm(0, hbf, wkl, kf, nullptr, nullptr, MT, Dc, Dc);
    gemm(0, hbf, wvl, vf, nullptr, nullptr, MT, Dc, Dc);
    k_rope_pack<<<dim3((Bc * Tc * NHc * HDc) / 256), b256, 0, stream>>>(
        qf, kf, vf, cosb, sinb, qb, kb2, vb2);
    k_attn<<<dim3(Bc * NHc * (Tc / 64)), dim3(128), 0, stream>>>(qb, kb2, vb2, ao);
    k_cvt_bf16<<<dim3((MT * Dc / 4) / 256), b256, 0, stream>>>(ao, aob, MT * Dc / 4);
    gemm(1, aob, wol, x, nullptr, nullptr, MT, Dc, Dc);   // residual add

    k_rmsnorm<true><<<dim3(MT), b256, 0, stream>>>(x, mlp_nw + (size_t)l * Dc, hbf);
    gemm(2, hbf, w1l, nullptr, sbf, nullptr, MT, Dc, DFFc);   // silu(u) -> bf16
    gemm(3, hbf, w3l, nullptr, gbf, sbf, MT, Dc, DFFc);       // g * silu(u) -> bf16
    gemm(1, gbf, w2l, x, nullptr, nullptr, MT, DFFc, Dc);     // residual add
  }
  k_rmsnorm<false><<<dim3(MT), b256, 0, stream>>>(x, fin_nw, d_out);
}
